// MoELayer_35656818492230
// MI455X (gfx1250) — compile-verified, matrix operand order checked by reading, not run on hardware
//
#include <hip/hip_runtime.h>
#include <math.h>

// MoE top-3 SwiGLU for MI455X (gfx1250), bf16 WMMA with fp32 accumulate.
// Workspace requirement: ~53 MB (bf16 copies of hs/w1/w3/w2 + routing lists).

#define NT   16384
#define HID  1024
#define FFN  256
#define NE   12
#define TOPK 3
#define PADLIST (NT*TOPK + NE*16)   // 49344: per-expert regions padded to 16
#define MAXTILES (PADLIST/16)       // 3084

typedef __attribute__((ext_vector_type(16))) __bf16 v16bf;
typedef __attribute__((ext_vector_type(8)))  float  v8f;

union FragU { int4 i4[2]; v16bf v; };

// Load one WMMA 16-bit A/B fragment: two contiguous 8-element (16B) runs,
// second run +16 elements, per the CDNA5 ISA VGPR striping.
__device__ __forceinline__ v16bf ldfrag(const __bf16* p) {
    FragU u;
    u.i4[0] = *(const int4*)(p);
    u.i4[1] = *(const int4*)(p + 16);
    return u.v;
}

__device__ __forceinline__ v8f zero8() {
    v8f z;
#pragma unroll
    for (int i = 0; i < 8; ++i) z[i] = 0.f;
    return z;
}

// ---------------------------------------------------------------- init ----
__global__ void init_kernel(float* __restrict__ finalp, int* __restrict__ tok,
                            float* __restrict__ wgt, int* __restrict__ tile_e,
                            int* __restrict__ counts, int* __restrict__ cursor,
                            float* __restrict__ importance, int* __restrict__ poffset) {
    size_t i = (size_t)blockIdx.x * blockDim.x + threadIdx.x;
    if (i < (size_t)NT * HID) finalp[i] = 0.f;
    if (i < PADLIST) { tok[i] = -1; wgt[i] = 0.f; }
    if (i < MAXTILES) tile_e[i] = -1;
    if (i < NE) { counts[i] = 0; cursor[i] = 0; importance[i] = 0.f; }
    if (i < NE + 1) poffset[i] = 0;
}

// ------------------------------------------------------------- convert ----
__global__ void cvt_kernel(const float* __restrict__ src, __bf16* __restrict__ dst, int n) {
    int i = blockIdx.x * blockDim.x + threadIdx.x;
    if (i < n) dst[i] = (__bf16)src[i];
}

// -------------------------------------------------------------- router ----
__global__ __launch_bounds__(128) void router_kernel(
    const float* __restrict__ hs, const float* __restrict__ gate,
    float* __restrict__ logits_out, float* __restrict__ sel_out,
    int* __restrict__ sel_e, float* __restrict__ sel_w,
    int* __restrict__ counts, float* __restrict__ importance) {
    __shared__ float gsh[NE * HID];               // 48 KB, broadcast gate weights
    for (int i = threadIdx.x; i < NE * HID; i += 128) gsh[i] = gate[i];
    __syncthreads();

    int wave = threadIdx.x >> 5, lane = threadIdx.x & 31;
    int token = blockIdx.x * 4 + wave;

    float acc[NE];
#pragma unroll
    for (int e = 0; e < NE; ++e) acc[e] = 0.f;
    const float* hrow = hs + (size_t)token * HID;
    for (int d = lane; d < HID; d += 32) {
        float x = hrow[d];
#pragma unroll
        for (int e = 0; e < NE; ++e) acc[e] += x * gsh[e * HID + d];
    }
#pragma unroll
    for (int e = 0; e < NE; ++e)
        for (int off = 16; off > 0; off >>= 1) acc[e] += __shfl_xor(acc[e], off, 32);

    if (lane == 0) {
        float mx = acc[0];
#pragma unroll
        for (int e = 1; e < NE; ++e) mx = fmaxf(mx, acc[e]);
        float p[NE], s = 0.f;
#pragma unroll
        for (int e = 0; e < NE; ++e) { p[e] = __expf(acc[e] - mx); s += p[e]; }
        float inv = 1.f / s;
#pragma unroll
        for (int e = 0; e < NE; ++e) { p[e] *= inv; logits_out[token * NE + e] = acc[e]; }

        int sel[TOPK]; float sw[TOPK]; bool used[NE];
#pragma unroll
        for (int e = 0; e < NE; ++e) used[e] = false;
        float tsum = 0.f;
#pragma unroll
        for (int k = 0; k < TOPK; ++k) {
            int best = 0; float bv = -1.f;
#pragma unroll
            for (int e = 0; e < NE; ++e)
                if (!used[e] && p[e] > bv) { bv = p[e]; best = e; }
            used[best] = true; sel[k] = best; sw[k] = bv; tsum += bv;
        }
        float invt = 1.f / tsum;
#pragma unroll
        for (int k = 0; k < TOPK; ++k) {
            float w = sw[k] * invt;
            sel_out[token * TOPK + k] = (float)sel[k];
            sel_e[token * TOPK + k] = sel[k];
            sel_w[token * TOPK + k] = w;
            atomicAdd(&counts[sel[k]], 1);
            atomicAdd(&importance[sel[k]], w);
        }
    }
}

// -------------------------------------------------------------- prefix ----
__global__ void prefix_kernel(const int* __restrict__ counts, int* __restrict__ poffset,
                              int* __restrict__ tile_e) {
    __shared__ int poff[NE + 1];
    if (threadIdx.x == 0) {
        int acc = 0;
        for (int e = 0; e < NE; ++e) {
            poff[e] = acc; poffset[e] = acc;
            acc += ((counts[e] + 15) >> 4) << 4;
        }
        poff[NE] = acc; poffset[NE] = acc;
    }
    __syncthreads();
    int ntiles = poff[NE] >> 4;
    for (int t = threadIdx.x; t < ntiles; t += blockDim.x) {
        int base = t << 4, e = 0;
        for (int j = 0; j < NE; ++j)
            if (base >= poff[j] && base < poff[j + 1]) e = j;
        tile_e[t] = e;
    }
}

// ------------------------------------------------------------- scatter ----
__global__ void scatter_kernel(const int* __restrict__ sel_e, const float* __restrict__ sel_w,
                               const int* __restrict__ poffset, int* __restrict__ cursor,
                               int* __restrict__ tok, float* __restrict__ wgt) {
    int i = blockIdx.x * blockDim.x + threadIdx.x;
    if (i >= NT * TOPK) return;
    int t = i / TOPK;
    int e = sel_e[i];
    float w = sel_w[i];
    int pos = atomicAdd(&cursor[e], 1);
    int idx = poffset[e] + pos;
    tok[idx] = t;
    wgt[idx] = w;
}

// ---------------------------------------------------- main expert GEMM ----
#define XSTR 1032   // 16B-aligned rows, +4 LDS banks/row -> conflict free b128
#define HSTR 264

__global__ __launch_bounds__(256) void moe_expert_kernel(
    const __bf16* __restrict__ hsb, const __bf16* __restrict__ w1b,
    const __bf16* __restrict__ w3b, const __bf16* __restrict__ w2b,
    const int* __restrict__ tile_e, const int* __restrict__ tok,
    const float* __restrict__ wgt, float* __restrict__ finalp) {
    int tile = blockIdx.x;
    int e = tile_e[tile];
    if (e < 0) return;                              // uniform per block

    __shared__ __align__(16) __bf16 Xs[16][XSTR];   // gathered X tile (bf16)
    __shared__ __align__(16) __bf16 Hs[16][HSTR];   // weighted SwiGLU output
    __shared__ int   stok[16];
    __shared__ float swgt[16];

    int tid = threadIdx.x, wave = tid >> 5, lane = tid & 31;
    int m = lane & 15, khalf = lane >> 4;

    if (tid < 16) {
        int t = tok[tile * 16 + tid];
        stok[tid] = t;
        swgt[tid] = (t >= 0) ? wgt[tile * 16 + tid] : 0.f;
    }
    __syncthreads();

    { // gather 16 token rows (bf16) into LDS; 16 threads per row, 8x b128 each
        int r = tid >> 4, c = tid & 15;
        int t = stok[r];
        int4* dst = (int4*)&Xs[r][c * 64];
        if (t >= 0) {
            const int4* src = (const int4*)(hsb + (size_t)t * HID + c * 64);
#pragma unroll
            for (int i = 0; i < 8; ++i) dst[i] = src[i];
        } else {
            int4 z; z.x = z.y = z.z = z.w = 0;
#pragma unroll
            for (int i = 0; i < 8; ++i) dst[i] = z;
        }
    }
    __syncthreads();

    // ---- phase 2: h1 = X*W1^T, h3 = X*W3^T (K=1024), 2 N-tiles per wave ----
    int nt0 = wave * 2, nt1 = nt0 + 1;
    const __bf16* w1r0 = w1b + ((size_t)e * FFN + nt0 * 16 + m) * HID + khalf * 8;
    const __bf16* w1r1 = w1b + ((size_t)e * FFN + nt1 * 16 + m) * HID + khalf * 8;
    const __bf16* w3r0 = w3b + ((size_t)e * FFN + nt0 * 16 + m) * HID + khalf * 8;
    const __bf16* w3r1 = w3b + ((size_t)e * FFN + nt1 * 16 + m) * HID + khalf * 8;

    v8f c10 = zero8(), c11 = zero8(), c30 = zero8(), c31 = zero8();
    for (int k0 = 0; k0 < HID; k0 += 32) {
        v16bf a   = ldfrag(&Xs[m][k0 + khalf * 8]);
        v16bf b10 = ldfrag(w1r0 + k0);
        v16bf b11 = ldfrag(w1r1 + k0);
        v16bf b30 = ldfrag(w3r0 + k0);
        v16bf b31 = ldfrag(w3r1 + k0);
        c10 = __builtin_amdgcn_wmma_f32_16x16x32_bf16(false, a, false, b10, (short)0, c10, false, false);
        c11 = __builtin_amdgcn_wmma_f32_16x16x32_bf16(false, a, false, b11, (short)0, c11, false, false);
        c30 = __builtin_amdgcn_wmma_f32_16x16x32_bf16(false, a, false, b30, (short)0, c30, false, false);
        c31 = __builtin_amdgcn_wmma_f32_16x16x32_bf16(false, a, false, b31, (short)0, c31, false, false);
    }
    // epilogue: silu(h1)*h3 * combine-weight -> Hs (bf16)
#pragma unroll
    for (int v = 0; v < 8; ++v) {
        int mr = v + khalf * 8;
        float w = swgt[mr];
        float x1 = c10[v], x3 = c30[v];
        Hs[mr][nt0 * 16 + m] = (__bf16)((x1 / (1.f + __expf(-x1))) * x3 * w);
        x1 = c11[v]; x3 = c31[v];
        Hs[mr][nt1 * 16 + m] = (__bf16)((x1 / (1.f + __expf(-x1))) * x3 * w);
    }
    __syncthreads();

    // ---- phase 3: out = H * W2^T (K=256, N=1024), 8 N-tiles per wave ----
    v16bf afr[8];
#pragma unroll
    for (int kk = 0; kk < 8; ++kk) afr[kk] = ldfrag(&Hs[m][kk * 32 + khalf * 8]);

    for (int nb = 0; nb < 8; ++nb) {
        int nt = wave * 8 + nb;
        int dcol = nt * 16 + m;
        const __bf16* w2r = w2b + ((size_t)e * HID + dcol) * FFN + khalf * 8;
        v8f c = zero8();
#pragma unroll
        for (int kk = 0; kk < 8; ++kk) {
            v16bf b = ldfrag(w2r + kk * 32);
            c = __builtin_amdgcn_wmma_f32_16x16x32_bf16(false, afr[kk], false, b, (short)0, c, false, false);
        }
#pragma unroll
        for (int v = 0; v < 8; ++v) {
            int mr = v + khalf * 8;
            int t = stok[mr];
            if (t >= 0) atomicAdd(&finalp[(size_t)t * HID + dcol], c[v]);
        }
    }
}

// ---------------------------------------------------------------- loss ----
__global__ void loss_kernel(const float* __restrict__ importance, float* __restrict__ out) {
    if (threadIdx.x == 0 && blockIdx.x == 0) {
        float mean = 0.f;
        for (int e = 0; e < NE; ++e) mean += importance[e];
        mean *= (1.f / NE);
        float var = 0.f;
        for (int e = 0; e < NE; ++e) { float d = importance[e] - mean; var += d * d; }
        var *= (1.f / (NE - 1));       // unbiased (ddof=1)
        out[0] = sqrtf(var);
    }
}

// -------------------------------------------------------------- launch ----
extern "C" void kernel_launch(void* const* d_in, const int* in_sizes, int n_in,
                              void* d_out, int out_size, void* d_ws, size_t ws_size,
                              hipStream_t stream) {
    const float* hs   = (const float*)d_in[0];
    const float* gate = (const float*)d_in[1];
    const float* w1   = (const float*)d_in[2];
    const float* w2   = (const float*)d_in[3];
    const float* w3   = (const float*)d_in[4];
    float* out = (float*)d_out;

    const size_t FINAL_N    = (size_t)NT * HID;
    const size_t LOGITS_OFF = FINAL_N;
    const size_t SEL_OFF    = LOGITS_OFF + (size_t)NT * NE;
    const size_t LOSS_OFF   = SEL_OFF + (size_t)NT * TOPK;

    char* ws = (char*)d_ws;
    size_t off = 0;
    auto take = [&](size_t bytes) -> char* {
        char* p = ws + off;
        off = (off + bytes + 255) & ~(size_t)255;
        return p;
    };
    __bf16* hsb  = (__bf16*)take((size_t)NT * HID * 2);
    __bf16* w1b  = (__bf16*)take((size_t)NE * FFN * HID * 2);
    __bf16* w3b  = (__bf16*)take((size_t)NE * FFN * HID * 2);
    __bf16* w2b  = (__bf16*)take((size_t)NE * HID * FFN * 2);
    int*   tokl  = (int*)take(PADLIST * 4);
    float* wgtl  = (float*)take(PADLIST * 4);
    int*   sel_e = (int*)take((size_t)NT * TOPK * 4);
    float* sel_w = (float*)take((size_t)NT * TOPK * 4);
    int*   tilee = (int*)take(MAXTILES * 4);
    int*   counts = (int*)take(64);
    int*   cursor = (int*)take(64);
    float* importance = (float*)take(64);
    int*   poffset = (int*)take(64);
    (void)ws_size; (void)in_sizes; (void)n_in; (void)out_size;

    init_kernel<<<(int)((FINAL_N + 255) / 256), 256, 0, stream>>>(
        out, tokl, wgtl, tilee, counts, cursor, importance, poffset);

    cvt_kernel<<<(NT * HID + 255) / 256, 256, 0, stream>>>(hs, hsb, NT * HID);
    const int wn = NE * FFN * HID;
    cvt_kernel<<<(wn + 255) / 256, 256, 0, stream>>>(w1, w1b, wn);
    cvt_kernel<<<(wn + 255) / 256, 256, 0, stream>>>(w3, w3b, wn);
    cvt_kernel<<<(wn + 255) / 256, 256, 0, stream>>>(w2, w2b, wn);

    router_kernel<<<NT / 4, 128, 0, stream>>>(hs, gate, out + LOGITS_OFF, out + SEL_OFF,
                                              sel_e, sel_w, counts, importance);
    prefix_kernel<<<1, 256, 0, stream>>>(counts, poffset, tilee);
    scatter_kernel<<<(NT * TOPK + 255) / 256, 256, 0, stream>>>(sel_e, sel_w, poffset,
                                                                cursor, tokl, wgtl);
    moe_expert_kernel<<<MAXTILES, 256, 0, stream>>>(hsb, w1b, w3b, w2b, tilee, tokl, wgtl, out);
    loss_kernel<<<1, 32, 0, stream>>>(importance, out + LOSS_OFF);
}
